// RNN_model_76622216560937
// MI455X (gfx1250) — compile-verified, matrix operand order checked by reading, not run on hardware
//
#include <hip/hip_runtime.h>
#include <hip/hip_bf16.h>
#include <cstdint>
#include <cstddef>

// Problem dims (match reference)
#define V_   32000
#define H_   1024
#define B_   64
#define T_   512
#define G4_  4096        // 4*H
#define EPS_ 1e-5f

typedef __attribute__((ext_vector_type(16))) _Float16 v16h;
typedef __attribute__((ext_vector_type(8)))  float    v8f;
typedef unsigned int u32x4 __attribute__((ext_vector_type(4)));
typedef int          i32x4 __attribute__((ext_vector_type(4)));
typedef int          i32x8 __attribute__((ext_vector_type(8)));

// ---------------------------------------------------------------------------
// Tile loader: 16 rows x 32 k of f16. Half-wave split: lanes 0-15 take
// k0..k0+15, lanes 16-31 take k0+16..k0+31 (contiguous k order; A and B tiles both use it).
// Works for global (vmem b128) and LDS (ds_load_b128) pointers.
// ---------------------------------------------------------------------------
__device__ __forceinline__ v16h ld_tile(const _Float16* base, int ld, int row0,
                                        int k0, int lane) {
  const int r  = row0 + (lane & 15);
  const int kk = k0 + ((lane >> 4) << 4);
  return *(const v16h*)(base + (size_t)r * ld + kk);
}

__device__ __forceinline__ float sigf(float x) { return 1.0f / (1.0f + __expf(-x)); }

// ---------------------------------------------------------------------------
// Tensor Data Mover: DMA a 2D f16 tile (width x height elems, row stride in
// elems) from global memory into LDS at byte offset lds_off. D# packed per
// CDNA5 ISA (group0: count/lds/global/type=2;
// group1: data_size=2B, tensor dims, tile dims, dim0 stride).
// ---------------------------------------------------------------------------
#if __has_builtin(__builtin_amdgcn_tensor_load_to_lds)
#define HAVE_TDM 1
__device__ __forceinline__ void tdm_load_2d_f16(unsigned lds_off,
                                                const _Float16* gptr,
                                                unsigned width, unsigned height,
                                                unsigned long long stride) {
  const unsigned long long ga = (unsigned long long)(uintptr_t)gptr;
  u32x4 g0;
  g0[0] = 1u;                                   // count=1 (valid), user mode
  g0[1] = lds_off;                              // lds_addr (bytes)
  g0[2] = (unsigned)(ga & 0xffffffffull);       // global_addr[31:0]
  g0[3] = (unsigned)((ga >> 32) & 0x01ffffffull) | 0x80000000u; // [56:32] | type=2
  i32x8 g1;
  g1[0] = (int)(1u << 16);                      // workgroup_mask=0, data_size=1 (2B)
  g1[1] = (int)((width & 0xffffu) << 16);       // tensor_dim0[15:0]   (bits 63:48)
  g1[2] = (int)((width >> 16) | ((height & 0xffffu) << 16)); // dim0 hi | dim1 lo
  g1[3] = (int)((height >> 16) | ((width & 0xffffu) << 16)); // dim1 hi | tile_dim0
  g1[4] = (int)(height & 0xffffu);              // tile_dim1 (tile_dim2 = 0)
  g1[5] = (int)(stride & 0xffffffffull);        // tensor_dim0_stride[31:0]
  g1[6] = (int)((stride >> 32) & 0xffffull);    // stride[47:32] | dim1_stride lo = 0
  g1[7] = 0;
  i32x4 z4 = {0, 0, 0, 0};
#if defined(__clang_major__) && (__clang_major__ >= 23)
  i32x8 z8 = {0, 0, 0, 0, 0, 0, 0, 0};
  __builtin_amdgcn_tensor_load_to_lds(g0, g1, z4, z4, z8, 0);
#else
  __builtin_amdgcn_tensor_load_to_lds(g0, g1, z4, z4, 0);
#endif
}
#else
#define HAVE_TDM 0
#endif

// Stage a [height x 1024] f16 activation slab into dynamic LDS (all waves must
// __syncthreads() after). Wave 0 issues the TDM op (EXEC ignored by TDM) and
// drains TENSORcnt before joining the barrier.
__device__ __forceinline__ void stage_slab(_Float16* lds_generic, unsigned lds_off,
                                           const _Float16* src, int height) {
#if HAVE_TDM
  if (threadIdx.x < 32) {
    tdm_load_2d_f16(lds_off, src, H_, (unsigned)height, H_);
    __builtin_amdgcn_s_wait_tensorcnt(0);
  }
  (void)lds_generic;
#else
  for (int i = threadIdx.x; i < height * H_; i += (int)blockDim.x)
    lds_generic[i] = src[i];
#endif
}

// ---------------------------------------------------------------------------
// f32 -> f16 elementwise convert (weights)
// ---------------------------------------------------------------------------
__global__ void k_f32_to_f16(const float* __restrict__ in,
                             _Float16* __restrict__ out, int n) {
  int i = blockIdx.x * blockDim.x + threadIdx.x;
  if (i < n) out[i] = (_Float16)in[i];
}

// ---------------------------------------------------------------------------
// Embedding gather + convert: E16[t*B + b][h] = f16(emb[x[b][t]][h])
// ---------------------------------------------------------------------------
__global__ void k_gather(const int* __restrict__ x,
                         const float* __restrict__ emb,
                         _Float16* __restrict__ e16) {
  const int r = blockIdx.x;            // r = t*B + b
  const int t = r / B_;
  const int b = r - t * B_;
  int idx = x[b * T_ + t];
  idx = (idx < 0) ? 0 : ((idx >= V_) ? (V_ - 1) : idx);
  const float* __restrict__ src = emb + (size_t)idx * H_;
  _Float16* __restrict__ dst = e16 + (size_t)r * H_;
  for (int j = threadIdx.x; j < H_; j += blockDim.x) dst[j] = (_Float16)src[j];
}

// ---------------------------------------------------------------------------
// State init (runs every call: determinism, ws is not re-poisoned)
// ---------------------------------------------------------------------------
__global__ void k_init(_Float16* h1, _Float16* h2, float* c1, float* c2,
                       float* pooled) {
  int i = blockIdx.x * blockDim.x + threadIdx.x;
  if (i < B_ * H_) {
    h1[i] = (_Float16)0.0f; h2[i] = (_Float16)0.0f;
    c1[i] = 0.0f; c2[i] = 0.0f;
    pooled[i] = -3.0e38f;
  }
}

// ---------------------------------------------------------------------------
// Big parallel WMMA GEMM: G1[r][n] = sum_k E16[r][k]*Wih1[n][k] + bias1[n].
// Block = 8 waves sharing a 64-row E slab (TDM-staged in LDS); each wave owns
// a 64(M)x16(N) tile strip. Fully-unrolled k-loop, branch-free, with operands
// for k+32 prefetched (A from LDS, B from global) before the k WMMAs issue.
// ---------------------------------------------------------------------------
__global__ __launch_bounds__(256) void k_gemm_xw1(
    const _Float16* __restrict__ E, const _Float16* __restrict__ W,
    const float* __restrict__ bih, const float* __restrict__ bhh,
    _Float16* __restrict__ G) {
  extern __shared__ _Float16 As[];                  // [64][1024] E slab
  const unsigned lds0 = __builtin_amdgcn_groupstaticsize();
  const int lane = threadIdx.x & 31;
  const int wv   = threadIdx.x >> 5;
  const int m0   = blockIdx.x * 64;
  const int n0   = blockIdx.y * 128 + wv * 16;
  const int ncol = n0 + (lane & 15);
  const float bias = bih[ncol] + bhh[ncol];

  stage_slab(As, lds0, E + (size_t)m0 * H_, 64);
  __syncthreads();

  v8f acc[4];
#pragma unroll
  for (int mt = 0; mt < 4; ++mt)
#pragma unroll
    for (int j = 0; j < 8; ++j) acc[mt][j] = bias;

  v16h bt = ld_tile(W, H_, n0, 0, lane);
  v16h at[4];
#pragma unroll
  for (int mt = 0; mt < 4; ++mt) at[mt] = ld_tile(As, H_, mt * 16, 0, lane);

#pragma unroll
  for (int k0 = 0; k0 < H_; k0 += 32) {
    const int kn = (k0 + 32) & (H_ - 1);    // last iter wraps to 0 (harmless)
    v16h btn = ld_tile(W, H_, n0, kn, lane);
    v16h atn[4];
#pragma unroll
    for (int mt = 0; mt < 4; ++mt) atn[mt] = ld_tile(As, H_, mt * 16, kn, lane);
#pragma unroll
    for (int mt = 0; mt < 4; ++mt)
      acc[mt] = __builtin_amdgcn_wmma_f32_16x16x32_f16(
          false, at[mt], false, bt, (short)0, acc[mt], false, false);
    bt = btn;
#pragma unroll
    for (int mt = 0; mt < 4; ++mt) at[mt] = atn[mt];
  }
  // C/D layout: VGPR j holds (m = j + 8*(lane>=16), n = lane&15)
  const int mh = (lane >> 4) * 8;
#pragma unroll
  for (int mt = 0; mt < 4; ++mt)
#pragma unroll
    for (int j = 0; j < 8; ++j) {
      const int m = m0 + mt * 16 + mh + j;
      G[(size_t)m * G4_ + ncol] = (_Float16)acc[mt][j];
    }
}

// ---------------------------------------------------------------------------
// Layer-1 recurrent step. One block per 32-feature chunk (32 blocks).
// gates = G1[t] (precomputed x-proj + bias) + h1_prev @ Whh1^T via WMMA,
// with h1_prev TDM-staged in LDS. Then LSTM nonlinearity, workgroup-local
// BatchNorm over the batch, locked-dropout mask.
// ---------------------------------------------------------------------------
__global__ __launch_bounds__(256) void k_step1(
    int t,
    const _Float16* __restrict__ G,     // [T*B,4096]
    const _Float16* __restrict__ Hr,    // [B,H] prev h (f16)
    const _Float16* __restrict__ Whh,   // [4096,1024] f16
    float* __restrict__ C,              // [B,H] cell state
    const float* __restrict__ gamma, const float* __restrict__ beta,
    const float* __restrict__ mask,     // [B,H]
    _Float16* __restrict__ Hw,          // [B,H] new h (f16)
    _Float16* __restrict__ O) {         // [B,H] BN(h)*mask (f16)
  __shared__ float gl[4][B_][32];
  __shared__ float hb[B_][32];
  __shared__ float mean_s[32], rstd_s[32];
  extern __shared__ _Float16 Hs[];                  // [64][1024] h slab
  const unsigned lds0 = __builtin_amdgcn_groupstaticsize();

  const int lane = threadIdx.x & 31;
  const int wv   = threadIdx.x >> 5;
  const int g    = wv & 3;              // gate block (i,f,g,o)
  const int nsub = (wv >> 2) * 16;      // 0 or 16 within the 32-feature chunk
  const int jg0  = blockIdx.x * 32;     // global feature base
  const int n0   = g * H_ + jg0 + nsub; // column base in 4096
  const int ncol = n0 + (lane & 15);
  const int mh   = (lane >> 4) * 8;

  stage_slab(Hs, lds0, Hr, B_);

  v8f acc[4];
  const size_t rbase = (size_t)t * B_;
#pragma unroll
  for (int mt = 0; mt < 4; ++mt)
#pragma unroll
    for (int j = 0; j < 8; ++j)
      acc[mt][j] = (float)G[(rbase + mt * 16 + mh + j) * G4_ + ncol];
  __syncthreads();

  v16h bt = ld_tile(Whh, H_, n0, 0, lane);
  v16h at[4];
#pragma unroll
  for (int mt = 0; mt < 4; ++mt) at[mt] = ld_tile(Hs, H_, mt * 16, 0, lane);

#pragma unroll
  for (int k0 = 0; k0 < H_; k0 += 32) {
    const int kn = (k0 + 32) & (H_ - 1);
    v16h btn = ld_tile(Whh, H_, n0, kn, lane);
    v16h atn[4];
#pragma unroll
    for (int mt = 0; mt < 4; ++mt) atn[mt] = ld_tile(Hs, H_, mt * 16, kn, lane);
#pragma unroll
    for (int mt = 0; mt < 4; ++mt)
      acc[mt] = __builtin_amdgcn_wmma_f32_16x16x32_f16(
          false, at[mt], false, bt, (short)0, acc[mt], false, false);
    bt = btn;
#pragma unroll
    for (int mt = 0; mt < 4; ++mt) at[mt] = atn[mt];
  }
#pragma unroll
  for (int mt = 0; mt < 4; ++mt)
#pragma unroll
    for (int j = 0; j < 8; ++j)
      gl[g][mt * 16 + mh + j][nsub + (lane & 15)] = acc[mt][j];
  __syncthreads();

  // LSTM nonlinearity (gate order i,f,g,o)
  for (int idx = threadIdx.x; idx < B_ * 32; idx += blockDim.x) {
    const int b = idx >> 5, j = idx & 31;
    const float gi = gl[0][b][j], gf = gl[1][b][j];
    const float gg = gl[2][b][j], go = gl[3][b][j];
    const int cidx = b * H_ + jg0 + j;
    const float c = sigf(gf) * C[cidx] + sigf(gi) * tanhf(gg);
    C[cidx] = c;
    const float h = sigf(go) * tanhf(c);
    hb[b][j] = h;
    Hw[cidx] = (_Float16)h;
  }
  __syncthreads();

  // BatchNorm stats over the batch (workgroup-local)
  if (threadIdx.x < 32) {
    const int j = threadIdx.x;
    float s = 0.0f, s2 = 0.0f;
    for (int b = 0; b < B_; ++b) { const float h = hb[b][j]; s += h; s2 += h * h; }
    const float m = s * (1.0f / B_);
    const float v = s2 * (1.0f / B_) - m * m;
    mean_s[j] = m;
    rstd_s[j] = rsqrtf(v + EPS_);
  }
  __syncthreads();

  for (int idx = threadIdx.x; idx < B_ * 32; idx += blockDim.x) {
    const int b = idx >> 5, j = idx & 31;
    const int jgl = jg0 + j;
    float o = gamma[jgl] * (hb[b][j] - mean_s[j]) * rstd_s[j] + beta[jgl];
    o *= mask[b * H_ + jgl];
    O[b * H_ + jgl] = (_Float16)o;
  }
}

// ---------------------------------------------------------------------------
// Layer-2 recurrent step: gates = bias2 + o1 @ Wih2^T + h2_prev @ Whh2^T.
// Both activation slabs (o1, h2) TDM-staged in LDS; both weight streams
// prefetched one k-iteration ahead. 8 WMMAs per k-step grouped so dependent
// accumulate pairs are 4 instructions apart. Then LSTM + BN + mask + running
// time-max into `pooled`.
// ---------------------------------------------------------------------------
__global__ __launch_bounds__(256) void k_step2(
    const _Float16* __restrict__ O1,    // [B,H] layer-1 output (f16)
    const _Float16* __restrict__ Hr,    // [B,H] prev h2 (f16)
    const _Float16* __restrict__ Wih, const _Float16* __restrict__ Whh,
    const float* __restrict__ bih, const float* __restrict__ bhh,
    float* __restrict__ C,
    const float* __restrict__ gamma, const float* __restrict__ beta,
    const float* __restrict__ mask,
    _Float16* __restrict__ Hw,
    float* __restrict__ pooled) {
  __shared__ float gl[4][B_][32];
  __shared__ float hb[B_][32];
  __shared__ float mean_s[32], rstd_s[32];
  extern __shared__ _Float16 Sl[];      // [64][1024] o1 slab, then [64][1024] h2 slab
  const unsigned lds0 = __builtin_amdgcn_groupstaticsize();
  _Float16* O1s = Sl;
  _Float16* H2s = Sl + (size_t)B_ * H_;

  const int lane = threadIdx.x & 31;
  const int wv   = threadIdx.x >> 5;
  const int g    = wv & 3;
  const int nsub = (wv >> 2) * 16;
  const int jg0  = blockIdx.x * 32;
  const int n0   = g * H_ + jg0 + nsub;
  const int ncol = n0 + (lane & 15);
  const int mh   = (lane >> 4) * 8;

  stage_slab(O1s, lds0, O1, B_);
  stage_slab(H2s, lds0 + (unsigned)(B_ * H_ * 2), Hr, B_);
  __syncthreads();

  const float bias = bih[ncol] + bhh[ncol];
  v8f acc[4];
#pragma unroll
  for (int mt = 0; mt < 4; ++mt)
#pragma unroll
    for (int j = 0; j < 8; ++j) acc[mt][j] = bias;

  v16h b1 = ld_tile(Wih, H_, n0, 0, lane);
  v16h b2 = ld_tile(Whh, H_, n0, 0, lane);
  v16h a1[4], a2[4];
#pragma unroll
  for (int mt = 0; mt < 4; ++mt) {
    a1[mt] = ld_tile(O1s, H_, mt * 16, 0, lane);
    a2[mt] = ld_tile(H2s, H_, mt * 16, 0, lane);
  }

#pragma unroll
  for (int k0 = 0; k0 < H_; k0 += 32) {
    const int kn = (k0 + 32) & (H_ - 1);
    v16h b1n = ld_tile(Wih, H_, n0, kn, lane);
    v16h b2n = ld_tile(Whh, H_, n0, kn, lane);
    v16h a1n[4], a2n[4];
#pragma unroll
    for (int mt = 0; mt < 4; ++mt) {
      a1n[mt] = ld_tile(O1s, H_, mt * 16, kn, lane);
      a2n[mt] = ld_tile(H2s, H_, mt * 16, kn, lane);
    }
#pragma unroll
    for (int mt = 0; mt < 4; ++mt)
      acc[mt] = __builtin_amdgcn_wmma_f32_16x16x32_f16(
          false, a1[mt], false, b1, (short)0, acc[mt], false, false);
#pragma unroll
    for (int mt = 0; mt < 4; ++mt)
      acc[mt] = __builtin_amdgcn_wmma_f32_16x16x32_f16(
          false, a2[mt], false, b2, (short)0, acc[mt], false, false);
    b1 = b1n; b2 = b2n;
#pragma unroll
    for (int mt = 0; mt < 4; ++mt) { a1[mt] = a1n[mt]; a2[mt] = a2n[mt]; }
  }
#pragma unroll
  for (int mt = 0; mt < 4; ++mt)
#pragma unroll
    for (int j = 0; j < 8; ++j)
      gl[g][mt * 16 + mh + j][nsub + (lane & 15)] = acc[mt][j];
  __syncthreads();

  for (int idx = threadIdx.x; idx < B_ * 32; idx += blockDim.x) {
    const int b = idx >> 5, j = idx & 31;
    const float gi = gl[0][b][j], gf = gl[1][b][j];
    const float gg = gl[2][b][j], go = gl[3][b][j];
    const int cidx = b * H_ + jg0 + j;
    const float c = sigf(gf) * C[cidx] + sigf(gi) * tanhf(gg);
    C[cidx] = c;
    const float h = sigf(go) * tanhf(c);
    hb[b][j] = h;
    Hw[cidx] = (_Float16)h;
  }
  __syncthreads();

  if (threadIdx.x < 32) {
    const int j = threadIdx.x;
    float s = 0.0f, s2 = 0.0f;
    for (int b = 0; b < B_; ++b) { const float h = hb[b][j]; s += h; s2 += h * h; }
    const float m = s * (1.0f / B_);
    const float v = s2 * (1.0f / B_) - m * m;
    mean_s[j] = m;
    rstd_s[j] = rsqrtf(v + EPS_);
  }
  __syncthreads();

  for (int idx = threadIdx.x; idx < B_ * 32; idx += blockDim.x) {
    const int b = idx >> 5, j = idx & 31;
    const int jgl = jg0 + j;
    float o = gamma[jgl] * (hb[b][j] - mean_s[j]) * rstd_s[j] + beta[jgl];
    o *= mask[b * H_ + jgl];
    const int pidx = b * H_ + jgl;
    pooled[pidx] = fmaxf(pooled[pidx], o);   // sole owner per element per step
  }
}

// ---------------------------------------------------------------------------
// Head: logits = pooled @ fc_w^T + fc_b ; loss = mean BCEWithLogits
// d_out = [loss, logits(64)]
// ---------------------------------------------------------------------------
__global__ __launch_bounds__(256) void k_final(
    const float* __restrict__ pooled, const float* __restrict__ fcw,
    const float* __restrict__ fcb, const float* __restrict__ tgt,
    float* __restrict__ out) {
  __shared__ float lg[B_];
  const int tid = threadIdx.x;
  if (tid < B_) {
    float s = 0.0f;
    const float* row = pooled + (size_t)tid * H_;
    for (int j = 0; j < H_; ++j) s += row[j] * fcw[j];
    lg[tid] = s + fcb[0];
  }
  __syncthreads();
  if (tid == 0) {
    float acc = 0.0f;
    for (int b = 0; b < B_; ++b) {
      const float l = lg[b];
      acc += fmaxf(l, 0.0f) - l * tgt[b] + log1pf(__expf(-fabsf(l)));
      out[1 + b] = l;
    }
    out[0] = acc * (1.0f / B_);
  }
}

// ---------------------------------------------------------------------------
// Host launcher
// ---------------------------------------------------------------------------
extern "C" void kernel_launch(void* const* d_in, const int* in_sizes, int n_in,
                              void* d_out, int out_size, void* d_ws, size_t ws_size,
                              hipStream_t stream) {
  const int*   x      = (const int*)  d_in[0];
  const float* tgt    = (const float*)d_in[1];
  const float* emb    = (const float*)d_in[2];
  const float* w_ih1  = (const float*)d_in[3];
  const float* w_hh1  = (const float*)d_in[4];
  const float* b_ih1  = (const float*)d_in[5];
  const float* b_hh1  = (const float*)d_in[6];
  const float* gamma1 = (const float*)d_in[7];
  const float* beta1  = (const float*)d_in[8];
  const float* w_ih2  = (const float*)d_in[9];
  const float* w_hh2  = (const float*)d_in[10];
  const float* b_ih2  = (const float*)d_in[11];
  const float* b_hh2  = (const float*)d_in[12];
  const float* gamma2 = (const float*)d_in[13];
  const float* beta2  = (const float*)d_in[14];
  const float* fc_w   = (const float*)d_in[15];
  const float* fc_b   = (const float*)d_in[16];
  const float* mask1  = (const float*)d_in[17];
  const float* mask2  = (const float*)d_in[18];

  char* ws = (char*)d_ws;
  size_t off = 0;
  auto carve = [&](size_t bytes) -> char* {
    char* p = ws + off;
    off += (bytes + 255) & ~(size_t)255;
    return p;
  };
  const size_t WN = (size_t)G4_ * H_;
  _Float16* Wih1h = (_Float16*)carve(WN * 2);
  _Float16* Whh1h = (_Float16*)carve(WN * 2);
  _Float16* Wih2h = (_Float16*)carve(WN * 2);
  _Float16* Whh2h = (_Float16*)carve(WN * 2);
  _Float16* E16   = (_Float16*)carve((size_t)T_ * B_ * H_ * 2);
  _Float16* G1    = (_Float16*)carve((size_t)T_ * B_ * G4_ * 2);
  _Float16* H1A   = (_Float16*)carve((size_t)B_ * H_ * 2);
  _Float16* H1B   = (_Float16*)carve((size_t)B_ * H_ * 2);
  _Float16* H2A   = (_Float16*)carve((size_t)B_ * H_ * 2);
  _Float16* H2B   = (_Float16*)carve((size_t)B_ * H_ * 2);
  _Float16* O1    = (_Float16*)carve((size_t)B_ * H_ * 2);
  float*    C1    = (float*)   carve((size_t)B_ * H_ * 4);
  float*    C2    = (float*)   carve((size_t)B_ * H_ * 4);
  float*    PO    = (float*)   carve((size_t)B_ * H_ * 4);

  const size_t SLAB = (size_t)B_ * H_ * sizeof(_Float16);   // 128 KB

  // 1) weight conversion f32 -> f16
  const int wn = (int)WN;
  k_f32_to_f16<<<(wn + 255) / 256, 256, 0, stream>>>(w_ih1, Wih1h, wn);
  k_f32_to_f16<<<(wn + 255) / 256, 256, 0, stream>>>(w_hh1, Whh1h, wn);
  k_f32_to_f16<<<(wn + 255) / 256, 256, 0, stream>>>(w_ih2, Wih2h, wn);
  k_f32_to_f16<<<(wn + 255) / 256, 256, 0, stream>>>(w_hh2, Whh2h, wn);

  // 2) embedding gather (+transpose to [T,B,H]) and state init
  k_gather<<<T_ * B_, 256, 0, stream>>>(x, emb, E16);
  k_init<<<(B_ * H_ + 255) / 256, 256, 0, stream>>>(H1A, H2A, C1, C2, PO);

  // 3) hoisted layer-1 input projection (big parallel WMMA GEMM, TDM-staged A)
  dim3 gg(T_ * B_ / 64, G4_ / 128);
  k_gemm_xw1<<<gg, 256, SLAB, stream>>>(E16, Wih1h, b_ih1, b_hh1, G1);

  // 4) sequential scan: 512 x (layer1 step, layer2 step), h ping-pong
  for (int t = 0; t < T_; ++t) {
    const _Float16* h1r = (t & 1) ? H1B : H1A;
    _Float16*       h1w = (t & 1) ? H1A : H1B;
    const _Float16* h2r = (t & 1) ? H2B : H2A;
    _Float16*       h2w = (t & 1) ? H2A : H2B;
    k_step1<<<H_ / 32, 256, SLAB, stream>>>(t, G1, h1r, Whh1h, C1,
                                            gamma1, beta1, mask1, h1w, O1);
    k_step2<<<H_ / 32, 256, 2 * SLAB, stream>>>(O1, h2r, Wih2h, Whh2h,
                                                b_ih2, b_hh2, C2, gamma2, beta2,
                                                mask2, h2w, PO);
  }

  // 5) head: logits + BCE loss
  k_final<<<1, 256, 0, stream>>>(PO, fc_w, fc_b, tgt, (float*)d_out);
}
